// Cell2Tissue_20667382629034
// MI455X (gfx1250) — matrix operand order repaired and checked by reading, not compiled, not run on hardware
//
#include <hip/hip_runtime.h>
#include <stdint.h>

typedef __attribute__((ext_vector_type(16))) __bf16 v16bf;
typedef __attribute__((ext_vector_type(8)))  float  v8f;

union FragU { uint32_t u[8]; v16bf v; };

__device__ __forceinline__ uint32_t f2bf(float f) {
  uint32_t u = __builtin_bit_cast(uint32_t, f);
  return (u + 0x7FFFu + ((u >> 16) & 1u)) >> 16;   // round-to-nearest-even bf16
}
__device__ __forceinline__ float bf2f(uint32_t h) {
  return __builtin_bit_cast(float, h << 16);
}

// ---------------------------------------------------------------------------
// Kernel 1: fold conv3x3 + avgpool4 into a 6x6/stride-4 conv; emit weights as
// hi/lo bf16 directly in WMMA B-fragment layout: [(kc*8+nt)*32+lane]*8+j dwords.
// B K order used here (contiguous, same as A): k = (lane>=16)*16 + 2*j + half, N = lane&15.
// ---------------------------------------------------------------------------
__global__ __launch_bounds__(256) void build_bfrag(
    const float* __restrict__ w, uint32_t* __restrict__ Bh,
    uint32_t* __restrict__ Bl) {
  unsigned tid  = blockIdx.x * 256u + threadIdx.x;   // < 144*8*32*8 = 294912
  unsigned j    = tid & 7u;
  unsigned lane = (tid >> 3) & 31u;
  unsigned nt   = (tid >> 8) & 7u;
  unsigned kc   = tid >> 11;                          // 0..143
  unsigned n    = nt * 16u + (lane & 15u);
  unsigned laneHi = lane >> 4;
  uint32_t dh = 0, dl = 0;
#pragma unroll
  for (int h = 0; h < 2; ++h) {
    unsigned klocal = laneHi * 16u + j * 2u + (unsigned)h;
    unsigned k  = kc * 32u + klocal;                  // 0..4607
    unsigned ic = k / 36u;
    unsigned t  = k - ic * 36u;                       // 0..35
    int u = (int)(t / 6u);
    int v = (int)(t - (unsigned)u * 6u);
    int ky0 = (u - 3 > 0) ? u - 3 : 0, ky1 = (u < 2) ? u : 2;
    int kx0 = (v - 3 > 0) ? v - 3 : 0, kx1 = (v < 2) ? v : 2;
    const float* wp = w + ((size_t)n * 128u + ic) * 9u;
    float s = 0.f;
    for (int ky = ky0; ky <= ky1; ++ky)
      for (int kx = kx0; kx <= kx1; ++kx)
        s += wp[ky * 3 + kx];
    s *= 0.0625f;                                     // fold 1/16 pool mean
    uint32_t hb = f2bf(s);
    dh |= hb << (16 * h);
    dl |= f2bf(s - bf2f(hb)) << (16 * h);             // lo correction term
  }
  Bh[tid] = dh;
  Bl[tid] = dl;
}

// ---------------------------------------------------------------------------
// Kernel 2: implicit GEMM  avg[M=4096, N=128] = A[M,K=4608] x Beff[K,N] + bias
// One wave per 16-row M tile; 8 N-tiles of 16; split-bf16: 3 WMMAs per tile.
// A K order used here (contiguous, same as B): k = (lane>=16)*16 + 2*j + half.
// ---------------------------------------------------------------------------
__global__ __launch_bounds__(32) void conv_pool_wmma(
    const float* __restrict__ cell, const uint32_t* __restrict__ Bh,
    const uint32_t* __restrict__ Bl, const float* __restrict__ bias,
    float* __restrict__ avg) {
  const int lane   = threadIdx.x & 31;
  const int mtile  = blockIdx.x;                      // 0..255
  const int laneHi = lane >> 4;
  const int mrow   = mtile * 16 + (lane & 15);
  const int p = mrow >> 6, q = mrow & 63;             // pooled (row, col)

  v8f acc[8];
#pragma unroll
  for (int nt = 0; nt < 8; ++nt)
#pragma unroll
    for (int r = 0; r < 8; ++r) acc[nt][r] = 0.f;

  for (int kc = 0; kc < 144; ++kc) {
    FragU ah, al;
#pragma unroll
    for (int j = 0; j < 8; ++j) {
      uint32_t dh = 0, dl = 0;
#pragma unroll
      for (int h = 0; h < 2; ++h) {
        int klocal = laneHi * 16 + j * 2 + h;
        int k  = kc * 32 + klocal;
        int ic = (int)((unsigned)k / 36u);
        int t  = k - ic * 36;
        int u  = (int)((unsigned)t / 6u);
        int v  = t - u * 6;
        int rr = 4 * p - 1 + u;                       // input row (SAME pad)
        int cc = 4 * q - 1 + v;                       // input col
        float val = 0.f;
        if ((unsigned)rr < 256u && (unsigned)cc < 256u)
          val = cell[(size_t)ic * 65536 + rr * 256 + cc];
        uint32_t hb = f2bf(val);
        dh |= hb << (16 * h);
        dl |= f2bf(val - bf2f(hb)) << (16 * h);
      }
      ah.u[j] = dh;
      al.u[j] = dl;
    }
#pragma unroll
    for (int nt = 0; nt < 8; ++nt) {
      const uint32_t* ph = Bh + ((size_t)(kc * 8 + nt) * 32 + lane) * 8;
      const uint32_t* pl = Bl + ((size_t)(kc * 8 + nt) * 32 + lane) * 8;
      FragU bh, bl;
#pragma unroll
      for (int j = 0; j < 8; ++j) { bh.u[j] = ph[j]; bl.u[j] = pl[j]; }
      // (Ah+Al)(Bh+Bl) ~= AhBh + AhBl + AlBh  (near-f32 precision on bf16 path)
      acc[nt] = __builtin_amdgcn_wmma_f32_16x16x32_bf16(
          false, ah.v, false, bh.v, (short)0, acc[nt], false, false);
      acc[nt] = __builtin_amdgcn_wmma_f32_16x16x32_bf16(
          false, ah.v, false, bl.v, (short)0, acc[nt], false, false);
      acc[nt] = __builtin_amdgcn_wmma_f32_16x16x32_bf16(
          false, al.v, false, bh.v, (short)0, acc[nt], false, false);
    }
  }
  // C/D layout: VGPR r -> M = r + 8*(lane>=16), N = nt*16 + (lane&15)
#pragma unroll
  for (int nt = 0; nt < 8; ++nt) {
    int n = nt * 16 + (lane & 15);
    float b = bias[n];
#pragma unroll
    for (int r = 0; r < 8; ++r) {
      int m = mtile * 16 + r + 8 * laneHi;
      avg[(size_t)n * 4096 + m] = acc[nt][r] + b;     // avg[c][p][q], m=p*64+q
    }
  }
}

// ---------------------------------------------------------------------------
// Kernel 3: out[i,j,c,y,x] = tissue[j,c,y,x] + (ROI_j ? avg[c,dy,dx] : 0),
// written to all 4 broadcast copies i=0..3 from one tissue read (L2-friendly).
// ---------------------------------------------------------------------------
__global__ __launch_bounds__(256) void assemble_out(
    const float* __restrict__ tissue, const float* __restrict__ avg,
    const int* __restrict__ loc, float* __restrict__ out) {
  unsigned idx = blockIdx.x * 256u + threadIdx.x;     // float4 units, < 8388608
  unsigned e = idx << 2;
  int x = e & 255;
  int y = (e >> 8) & 255;
  int c = (e >> 16) & 127;
  int j = e >> 23;
  float4 v = reinterpret_cast<const float4*>(tissue)[idx];
  int rs = (loc[2 * j + 1] >> 2) - 32;                // row start from loc[:,1]
  int cs = (loc[2 * j + 0] >> 2) - 32;                // col start from loc[:,0]
  int dy = y - rs;
  if ((unsigned)dy < 64u) {
    const float* arow = avg + (size_t)c * 4096 + dy * 64;
    int dx = x - cs;
    float* vp = &v.x;
#pragma unroll
    for (int m = 0; m < 4; ++m) {
      int d = dx + m;
      if ((unsigned)d < 64u) vp[m] += arow[d];
    }
  }
  float4* o = reinterpret_cast<float4*>(out);
  o[idx]             = v;
  o[idx +  8388608u] = v;
  o[idx + 16777216u] = v;
  o[idx + 25165824u] = v;
}

extern "C" void kernel_launch(void* const* d_in, const int* in_sizes, int n_in,
                              void* d_out, int out_size, void* d_ws,
                              size_t ws_size, hipStream_t stream) {
  const float* tissue = (const float*)d_in[0];  // (4,128,256,256)
  const float* cell   = (const float*)d_in[1];  // (1,128,256,256)
  const int*   loc    = (const int*)d_in[2];    // (4,2)
  const float* conv_w = (const float*)d_in[3];  // (128,128,3,3)
  const float* conv_b = (const float*)d_in[4];  // (128,)
  float* out = (float*)d_out;

  char* ws = (char*)d_ws;
  uint32_t* Bh  = (uint32_t*)ws;                 // 294912 dwords = 1.125 MB
  uint32_t* Bl  = (uint32_t*)(ws + 1179648);     // 1.125 MB
  float*    avg = (float*)(ws + 2359296);        // 128*64*64 f32 = 2 MB

  build_bfrag<<<1152, 256, 0, stream>>>(conv_w, Bh, Bl);
  conv_pool_wmma<<<256, 32, 0, stream>>>(cell, Bh, Bl, conv_b, avg);
  assemble_out<<<32768, 256, 0, stream>>>(tissue, avg, loc, out);
}
